// BucketSwinAttention_12627203851167
// MI455X (gfx1250) — compile-verified, matrix operand order checked by reading, not run on hardware
//
#include <hip/hip_runtime.h>

// Bucketed local attention for MI455X (gfx1250), wave32 + WMMA f16->f32.
// One workgroup (8 wave32) per (batch, bucket, head); each wave owns 16 query rows.
// Streaming global traffic uses non-temporal hints (zero reuse; don't pollute L2).

typedef _Float16 h1;
typedef __attribute__((ext_vector_type(4)))  _Float16 h4v;
typedef __attribute__((ext_vector_type(8)))  _Float16 h8v;
typedef __attribute__((ext_vector_type(16))) _Float16 v16h;
typedef __attribute__((ext_vector_type(4)))  float    f4v;   // native vector: OK for NT builtins
typedef __attribute__((ext_vector_type(8)))  float    v8f;

#define Bn   2
#define Ln   65536
#define Hn   8
#define Dn   32
#define BUCK 128
#define NBK  (Ln / BUCK)
#define ROWF (Hn * Dn)   // 256 floats between consecutive tokens

// LDS row strides in halfs (padded; all chunk offsets stay 16B aligned)
#define KS 40
#define VS 136
#define PS 136

// Load 16 halfs in the A/B per-lane pattern: two contiguous 8-half (16B) chunks
// at K-offsets c0 and c0+16 (c0 = 0 for lanes 0-15, 8 for lanes 16-31).
__device__ __forceinline__ v16h lds_load_16h(const h1* rowbase, int c0) {
    h8v lo = *(const h8v*)(rowbase + c0);
    h8v hi = *(const h8v*)(rowbase + c0 + 16);
    v16h r;
#pragma unroll
    for (int i = 0; i < 8; ++i) { r[i] = lo[i]; r[8 + i] = hi[i]; }
    return r;
}

__global__ __launch_bounds__(256) void bucket_attn_wmma(
    const float* __restrict__ Q, const float* __restrict__ K,
    const float* __restrict__ V, const int* __restrict__ scope,
    float* __restrict__ Out)
{
    __shared__ h1 Ksh[BUCK * KS];      // K row-major  [token][d]
    __shared__ h1 Vsh[Dn * VS];        // V transposed [d][token]
    __shared__ h1 Psh[8 * 16 * PS];    // per-wave P spill [wave*16 + m][k]

    const int blk  = blockIdx.x;
    const int h    = blk % Hn;
    const int nb   = (blk / Hn) % NBK;
    const int b    = blk / (Hn * NBK);

    const int tid  = threadIdx.x;
    const int wave = tid >> 5;
    const int lane = tid & 31;
    const int ln16 = lane & 15;
    const int hi16 = lane >> 4;        // which 16-lane half
    const int c0   = hi16 * 8;         // K-dim chunk base for A/B layouts

    const int tok0  = nb * BUCK;
    const int start = scope[(b * NBK + nb) * 2 + 0];
    const int end   = scope[(b * NBK + nb) * 2 + 1];

    const size_t base = ((size_t)(b * Ln + tok0) * Hn + h) * (size_t)Dn;
    const float* Kg = K + base;
    const float* Vg = V + base;

    // ---- cooperative stage: K (row-major f16), V (transposed f16) ----
#pragma unroll
    for (int it = 0; it < (BUCK * Dn / 4) / 256; ++it) {
        int i     = it * 256 + tid;
        int token = i >> 3;            // 8 float4 per 32-float row
        int dc    = (i & 7) << 2;
        f4v kv = __builtin_nontemporal_load(
            (const f4v*)(Kg + (size_t)token * ROWF + dc));
        f4v vv = __builtin_nontemporal_load(
            (const f4v*)(Vg + (size_t)token * ROWF + dc));
        h4v k4 = { (h1)kv[0], (h1)kv[1], (h1)kv[2], (h1)kv[3] };
        *(h4v*)(Ksh + token * KS + dc) = k4;
        Vsh[(dc + 0) * VS + token] = (h1)vv[0];
        Vsh[(dc + 1) * VS + token] = (h1)vv[1];
        Vsh[(dc + 2) * VS + token] = (h1)vv[2];
        Vsh[(dc + 3) * VS + token] = (h1)vv[3];
    }

    // ---- load this wave's 16x32 Q tile from global in A layout (pre-barrier,
    //      so the b128 NT loads overlap the staging barrier) ----
    const int qrow = wave * 16 + ln16;          // row within bucket
    const float* qp = Q + base + (size_t)qrow * ROWF;
    f4v q0 = __builtin_nontemporal_load((const f4v*)(qp + c0));
    f4v q1 = __builtin_nontemporal_load((const f4v*)(qp + c0 + 4));
    f4v q2 = __builtin_nontemporal_load((const f4v*)(qp + c0 + 16));
    f4v q3 = __builtin_nontemporal_load((const f4v*)(qp + c0 + 20));
    v16h aq;
#pragma unroll
    for (int i = 0; i < 4; ++i) {
        aq[i]      = (h1)q0[i];
        aq[4 + i]  = (h1)q1[i];
        aq[8 + i]  = (h1)q2[i];
        aq[12 + i] = (h1)q3[i];
    }

    __syncthreads();

    // ---- S = Q K^T : 8 score tiles, one 16x16x32 WMMA each (K=32=D) ----
    v8f acc[8];
    const v8f zero = {};
#pragma unroll
    for (int t = 0; t < 8; ++t) {
        const h1* krow = Ksh + (t * 16 + ln16) * KS;   // B col = key = t*16+ln16
        v16h bk = lds_load_16h(krow, c0);
        acc[t] = __builtin_amdgcn_wmma_f32_16x16x32_f16(
            false, aq, false, bk, (short)0, zero, false, false);
    }

    // ---- masked row softmax (f32). C/D layout: lane holds N=ln16, M=r+8*hi16 ----
    const float scale = 0.1767766952966369f;   // 1/sqrt(32)
    bool colOk[8];
#pragma unroll
    for (int t = 0; t < 8; ++t) {
        int col = tok0 + t * 16 + ln16;
        colOk[t] = (col >= start) && (col < end);
    }
    float rinv[8];
#pragma unroll
    for (int r = 0; r < 8; ++r) {
        float mx = -3.0e38f;
#pragma unroll
        for (int t = 0; t < 8; ++t) {
            float s = colOk[t] ? acc[t][r] * scale : -1.0e30f;
            acc[t][r] = s;
            mx = fmaxf(mx, s);
        }
#pragma unroll
        for (int m = 1; m <= 8; m <<= 1) mx = fmaxf(mx, __shfl_xor(mx, m, 32));
        float sum = 0.f;
#pragma unroll
        for (int t = 0; t < 8; ++t) {
            float e = __expf(acc[t][r] - mx);
            acc[t][r] = e;
            sum += e;
        }
#pragma unroll
        for (int m = 1; m <= 8; m <<= 1) sum += __shfl_xor(sum, m, 32);
        rinv[r] = 1.0f / sum;
    }

    // ---- spill P (f16) to the wave-private LDS strip, reload in A layout.
    //      LDS ops from one wave execute in order (ISA 7.3): no barrier needed,
    //      the compiler's s_wait_dscnt covers the RAW on the reloads. ----
    h1* pw = Psh + wave * 16 * PS;
#pragma unroll
    for (int t = 0; t < 8; ++t)
#pragma unroll
        for (int r = 0; r < 8; ++r)
            pw[(r + 8 * hi16) * PS + t * 16 + ln16] = (h1)acc[t][r];

    // ---- O = P V : 4 k-chunks of 32, two 16-wide d tiles ----
    v8f o0 = zero, o1 = zero;
    const h1* prow = pw + ln16 * PS;            // A row = ln16 (both lane halves)
#pragma unroll
    for (int c = 0; c < 4; ++c) {
        v16h ap  = lds_load_16h(prow + c * 32, c0);
        v16h bv0 = lds_load_16h(Vsh + (0  + ln16) * VS + c * 32, c0); // B col=d
        v16h bv1 = lds_load_16h(Vsh + (16 + ln16) * VS + c * 32, c0);
        o0 = __builtin_amdgcn_wmma_f32_16x16x32_f16(
            false, ap, false, bv0, (short)0, o0, false, false);
        o1 = __builtin_amdgcn_wmma_f32_16x16x32_f16(
            false, ap, false, bv1, (short)0, o1, false, false);
    }

    // ---- normalize, zero out-of-scope queries, store f32 (non-temporal) ----
    float* ob = Out + base + (size_t)(wave * 16) * ROWF;
#pragma unroll
    for (int r = 0; r < 8; ++r) {
        int m  = r + 8 * hi16;
        int qa = tok0 + wave * 16 + m;
        float sc = ((qa >= start) && (qa < end)) ? rinv[r] : 0.f;
        float* orow = ob + (size_t)m * ROWF;
        __builtin_nontemporal_store(o0[r] * sc, orow + ln16);
        __builtin_nontemporal_store(o1[r] * sc, orow + 16 + ln16);
    }
}

extern "C" void kernel_launch(void* const* d_in, const int* in_sizes, int n_in,
                              void* d_out, int out_size, void* d_ws, size_t ws_size,
                              hipStream_t stream) {
    (void)in_sizes; (void)n_in; (void)out_size; (void)d_ws; (void)ws_size;
    const float* Q = (const float*)d_in[0];
    const float* K = (const float*)d_in[1];
    const float* V = (const float*)d_in[2];
    const int* scope = (const int*)d_in[3];   // [B, nb, 2]
    // d_in[4] = buck_size scalar (hardcoded 128 to match reference setup)
    float* Out = (float*)d_out;

    dim3 grid(Bn * NBK * Hn);   // 8192 blocks: one per (b, bucket, head)
    dim3 block(256);            // 8 wave32
    bucket_attn_wmma<<<grid, block, 0, stream>>>(Q, K, V, scope, Out);
}
